// AGNNNet_84774064488691
// MI455X (gfx1250) — matrix operand order repaired and checked by reading, not compiled, not run on hardware
//
#include <hip/hip_runtime.h>
#include <math.h>

typedef __attribute__((ext_vector_type(16))) __bf16 v16bf;
typedef __attribute__((ext_vector_type(8)))  float  v8f;

#define HID 64
#define EPS 1e-12f

// ---------- small utility kernels ----------
__global__ void fill_f32_kernel(float* p, float v, long long n) {
  long long i = (long long)blockIdx.x * blockDim.x + threadIdx.x;
  if (i < n) p[i] = v;
}
__global__ void fill_u32_kernel(unsigned* p, unsigned v, long long n) {
  long long i = (long long)blockIdx.x * blockDim.x + threadIdx.x;
  if (i < n) p[i] = v;
}
__global__ void cvt_bf16_kernel(const float* __restrict__ s, __bf16* __restrict__ d, long long n) {
  long long i = (long long)blockIdx.x * blockDim.x + threadIdx.x;
  if (i < n) d[i] = (__bf16)s[i];
}

// order-preserving float <-> uint encoding for atomic max
__device__ __forceinline__ unsigned fenc(float f) {
  unsigned u = __float_as_uint(f);
  return (u & 0x80000000u) ? ~u : (u | 0x80000000u);
}
__device__ __forceinline__ float fdec(unsigned u) {
  u = (u & 0x80000000u) ? (u & 0x7fffffffu) : ~u;
  return __uint_as_float(u);
}

// ---------- WMMA GEMM: one wave computes 16 rows x 64 cols of OUT = X(f32)*W^T(bf16)+bias ----------
__device__ __forceinline__ void wave_gemm_tile(
    const float* __restrict__ X, const __bf16* __restrict__ W,
    const float* __restrict__ bias, float* __restrict__ OUT,
    int Nrows, int K, int ldout, int m0, int nbase, bool relu, int lane) {
  const int M = lane & 15, half = lane >> 4;
  int row = m0 + M; if (row >= Nrows) row = Nrows - 1;
  const float* xrow = X + (size_t)row * K;

  v8f acc[4];
#pragma unroll
  for (int t = 0; t < 4; ++t) acc[t] = (v8f){};

  const __bf16* wlane = W + (size_t)(nbase + (lane & 15)) * K + (size_t)half * 16;

  for (int k0 = 0; k0 < K; k0 += 32) {
    // A fragment: 16 contiguous K values (k = 16*half + s), matching B
    const float* p = xrow + k0 + half * 16;
    float4 f0 = *(const float4*)(p);
    float4 f1 = *(const float4*)(p + 4);
    float4 f2 = *(const float4*)(p + 8);
    float4 f3 = *(const float4*)(p + 12);
    v16bf a;
    a[0]=(__bf16)f0.x;  a[1]=(__bf16)f0.y;  a[2]=(__bf16)f0.z;  a[3]=(__bf16)f0.w;
    a[4]=(__bf16)f1.x;  a[5]=(__bf16)f1.y;  a[6]=(__bf16)f1.z;  a[7]=(__bf16)f1.w;
    a[8]=(__bf16)f2.x;  a[9]=(__bf16)f2.y;  a[10]=(__bf16)f2.z; a[11]=(__bf16)f2.w;
    a[12]=(__bf16)f3.x; a[13]=(__bf16)f3.y; a[14]=(__bf16)f3.z; a[15]=(__bf16)f3.w;

#pragma unroll
    for (int t = 0; t < 4; ++t) {
      // B fragment: 16 contiguous K values of W row (nbase + 16t + (lane&15))
      v16bf b = *(const v16bf*)(wlane + (size_t)(16 * t) * K + k0);
      acc[t] = __builtin_amdgcn_wmma_f32_16x16x32_bf16(
          false, a, false, b, (short)0, acc[t], false, false);
    }
  }

#pragma unroll
  for (int t = 0; t < 4; ++t) {
    float bv = bias[nbase + t * 16 + (lane & 15)];
#pragma unroll
    for (int r = 0; r < 8; ++r) {
      int rr = m0 + r + 8 * half;
      if (rr < Nrows) {
        float v = acc[t][r] + bv;
        if (relu) v = fmaxf(v, 0.f);
        OUT[(size_t)rr * ldout + nbase + t * 16 + (lane & 15)] = v;
      }
    }
  }
}

// GEMM1: H[N,64] = relu(X[N,K] @ W1^T + b1); one wave per 16-row tile, all 64 cols
__global__ void gemm1_kernel(const float* __restrict__ X, const __bf16* __restrict__ W,
                             const float* __restrict__ bias, float* __restrict__ H,
                             int Nrows, int K) {
  int wid  = (blockIdx.x * blockDim.x + threadIdx.x) >> 5;
  int lane = threadIdx.x & 31;
  int mmax = (Nrows + 15) >> 4;
  if (wid >= mmax) return;  // wave-uniform exit keeps EXEC all-1s for WMMA
  wave_gemm_tile(X, W, bias, H, Nrows, K, HID, wid * 16, 0, true, lane);
}

// GEMM2: LOGITS[N,C] = H2[N,64] @ W2^T + b2; block = C/64 waves, one 16-row tile per block
__global__ void gemm2_kernel(const float* __restrict__ X, const __bf16* __restrict__ W,
                             const float* __restrict__ bias, float* __restrict__ L,
                             int Nrows, int C) {
  int wid  = threadIdx.x >> 5;
  int lane = threadIdx.x & 31;
  wave_gemm_tile(X, W, bias, L, Nrows, HID, C, blockIdx.x * 16, wid * 64, false, lane);
}

// ---------- AGNN edge kernels ----------
__global__ void norm_kernel(const float* __restrict__ h, float* __restrict__ nrm, int N) {
  int i = blockIdx.x * blockDim.x + threadIdx.x;
  if (i >= N) return;
  const float* p = h + (size_t)i * HID;
  float s = 0.f;
#pragma unroll
  for (int j = 0; j < HID; j += 4) {
    float4 f = *(const float4*)(p + j);
    s += f.x * f.x + f.y * f.y + f.z * f.z + f.w * f.w;
  }
  nrm[i] = fmaxf(sqrtf(s), EPS);
}

// 8 lanes per edge: a[e] = beta * dot(h[dst],h[src]) / (norm[src]*norm[dst]); atomic max into amax[dst]
__global__ void edge_dot_kernel(const float* __restrict__ h, const float* __restrict__ nrm,
                                const long long* __restrict__ ei, long long E,
                                const float* __restrict__ beta,
                                float* __restrict__ a_out, unsigned* __restrict__ amax) {
  long long t = (long long)blockIdx.x * blockDim.x + threadIdx.x;
  long long e = t >> 3;
  int l8 = threadIdx.x & 7;
  if (e >= E) return;
  long long s = ei[e], d = ei[E + e];
  const float* hs = h + (size_t)s * HID + l8 * 8;
  const float* hd = h + (size_t)d * HID + l8 * 8;
  float4 a0 = *(const float4*)(hs),     a1 = *(const float4*)(hs + 4);
  float4 b0 = *(const float4*)(hd),     b1 = *(const float4*)(hd + 4);
  float acc = a0.x*b0.x + a0.y*b0.y + a0.z*b0.z + a0.w*b0.w
            + a1.x*b1.x + a1.y*b1.y + a1.z*b1.z + a1.w*b1.w;
  acc += __shfl_xor(acc, 4, 8);
  acc += __shfl_xor(acc, 2, 8);
  acc += __shfl_xor(acc, 1, 8);
  if (l8 == 0) {
    float av = beta[0] * acc / (nrm[s] * nrm[d]);
    a_out[e] = av;
    atomicMax(&amax[d], fenc(av));
  }
}

// thread per edge: aexp[e] = exp(a - amax[dst]); denom[dst] += aexp
__global__ void edge_exp_kernel(const long long* __restrict__ ei, long long E,
                                float* __restrict__ a_buf, const unsigned* __restrict__ amax,
                                float* __restrict__ denom) {
  long long e = (long long)blockIdx.x * blockDim.x + threadIdx.x;
  if (e >= E) return;
  long long d = ei[E + e];
  float ev = expf(a_buf[e] - fdec(amax[d]));
  a_buf[e] = ev;
  atomicAdd(&denom[d], ev);
}

// 8 lanes per edge: hout[dst] += (aexp[e]/denom[dst]) * h[src]
__global__ void edge_aggr_kernel(const float* __restrict__ h, const long long* __restrict__ ei,
                                 long long E, const float* __restrict__ aexp,
                                 const float* __restrict__ denom, float* __restrict__ hout) {
  long long t = (long long)blockIdx.x * blockDim.x + threadIdx.x;
  long long e = t >> 3;
  int l8 = threadIdx.x & 7;
  if (e >= E) return;
  long long s = ei[e], d = ei[E + e];
  float alpha = aexp[e] / fmaxf(denom[d], EPS);
  const float* hs = h + (size_t)s * HID + l8 * 8;
  float* od = hout + (size_t)d * HID + l8 * 8;
#pragma unroll
  for (int j = 0; j < 8; ++j) atomicAdd(&od[j], alpha * hs[j]);
}

// ---------- log-softmax in place on d_out, one block (C threads) per row ----------
__global__ void logsoftmax_kernel(float* __restrict__ out, int C) {
  __shared__ float red[256];
  int row = blockIdx.x, tid = threadIdx.x;
  float v = out[(size_t)row * C + tid];
  red[tid] = v; __syncthreads();
  for (int s = blockDim.x >> 1; s > 0; s >>= 1) {
    if (tid < s) red[tid] = fmaxf(red[tid], red[tid + s]);
    __syncthreads();
  }
  float m = red[0]; __syncthreads();
  float ex = expf(v - m);
  red[tid] = ex; __syncthreads();
  for (int s = blockDim.x >> 1; s > 0; s >>= 1) {
    if (tid < s) red[tid] += red[tid + s];
    __syncthreads();
  }
  out[(size_t)row * C + tid] = v - m - logf(red[0]);
}

// ---------- host-side orchestration ----------
static void agnn_layer(const float* h_in, float* h_out, const float* beta,
                       const long long* ei, long long E, int N,
                       float* nrm, unsigned* amax, float* denom, float* aexp,
                       hipStream_t stream) {
  long long nh = (long long)N * HID;
  fill_f32_kernel<<<(unsigned)((nh + 255) / 256), 256, 0, stream>>>(h_out, 0.f, nh);
  fill_f32_kernel<<<(unsigned)((N + 255) / 256), 256, 0, stream>>>(denom, 0.f, N);
  fill_u32_kernel<<<(unsigned)((N + 255) / 256), 256, 0, stream>>>(amax, 0u, N);
  norm_kernel<<<(unsigned)((N + 255) / 256), 256, 0, stream>>>(h_in, nrm, N);
  long long et = E * 8;
  edge_dot_kernel<<<(unsigned)((et + 255) / 256), 256, 0, stream>>>(h_in, nrm, ei, E, beta, aexp, amax);
  edge_exp_kernel<<<(unsigned)((E + 255) / 256), 256, 0, stream>>>(ei, E, aexp, amax, denom);
  edge_aggr_kernel<<<(unsigned)((et + 255) / 256), 256, 0, stream>>>(h_in, ei, E, aexp, denom, h_out);
}

extern "C" void kernel_launch(void* const* d_in, const int* in_sizes, int n_in,
                              void* d_out, int out_size, void* d_ws, size_t ws_size,
                              hipStream_t stream) {
  const float*     x     = (const float*)d_in[0];
  const long long* ei    = (const long long*)d_in[1];
  const float*     w1    = (const float*)d_in[2];
  const float*     b1    = (const float*)d_in[3];
  const float*     beta1 = (const float*)d_in[4];
  const float*     beta2 = (const float*)d_in[5];
  const float*     w2    = (const float*)d_in[6];
  const float*     b2    = (const float*)d_in[7];

  const int Fin = in_sizes[2] / HID;        // 1024
  const int N   = in_sizes[0] / Fin;        // 50000
  const long long E = in_sizes[1] / 2;      // 800000
  const int C   = in_sizes[6] / HID;        // 256

  char* ws = (char*)d_ws;
  size_t off = 0;
  auto carve = [&](size_t bytes) { size_t o = off; off = (off + bytes + 255) & ~(size_t)255; return o; };
  __bf16*   w1bf  = (__bf16*)(ws + carve((size_t)HID * Fin * 2));
  __bf16*   w2bf  = (__bf16*)(ws + carve((size_t)C * HID * 2));
  float*    h0    = (float*)(ws + carve((size_t)N * HID * 4));
  float*    h1    = (float*)(ws + carve((size_t)N * HID * 4));
  float*    h2    = (float*)(ws + carve((size_t)N * HID * 4));
  float*    nrm   = (float*)(ws + carve((size_t)N * 4));
  unsigned* amax  = (unsigned*)(ws + carve((size_t)N * 4));
  float*    denom = (float*)(ws + carve((size_t)N * 4));
  float*    aexp  = (float*)(ws + carve((size_t)E * 4));

  // convert weights fp32 -> bf16 (row-major; WMMA B fragments read contiguous K)
  long long nw1 = (long long)HID * Fin, nw2 = (long long)C * HID;
  cvt_bf16_kernel<<<(unsigned)((nw1 + 255) / 256), 256, 0, stream>>>(w1, w1bf, nw1);
  cvt_bf16_kernel<<<(unsigned)((nw2 + 255) / 256), 256, 0, stream>>>(w2, w2bf, nw2);

  // GEMM1 + ReLU: h0 = relu(x @ w1^T + b1)
  int mmax = (N + 15) / 16;
  gemm1_kernel<<<(mmax + 3) / 4, 128, 0, stream>>>(x, w1bf, b1, h0, N, Fin);

  // two AGNN propagation layers
  agnn_layer(h0, h1, beta1, ei, E, N, nrm, amax, denom, aexp, stream);
  agnn_layer(h1, h2, beta2, ei, E, N, nrm, amax, denom, aexp, stream);

  // GEMM2: logits = h2 @ w2^T + b2, written straight into d_out
  float* logits = (float*)d_out;
  gemm2_kernel<<<mmax, 32 * (C / 64), 0, stream>>>(h2, w2bf, b2, logits, N, C);

  // log-softmax in place on d_out
  logsoftmax_kernel<<<N, C, 0, stream>>>(logits, C);
}
